// Attention_28819230556655
// MI455X (gfx1250) — matrix operand order repaired and checked by reading, not compiled, not run on hardware
//
#include <hip/hip_runtime.h>
#include <cmath>
#include <cstdint>

typedef _Float16 f16;
typedef __attribute__((ext_vector_type(16))) _Float16 v16h;
typedef __attribute__((ext_vector_type(8)))  _Float16 v8h;
typedef __attribute__((ext_vector_type(8)))  float    v8f;

#define B_  2
#define S_  2048
#define D_  4096
#define H_  32
#define HD_ 128
#define AL_ 10

static __device__ __forceinline__ v8f wmma_f16(v16h a, v16h b, v8f c) {
  // D = A(16x32) * B(32x16) + C(16x16 f32)
  return __builtin_amdgcn_wmma_f32_16x16x32_f16(false, a, false, b, (short)0, c,
                                                false, false);
}

// Async copy of 2x16B (32 bytes) global -> LDS, tracked by ASYNCcnt.
static __device__ __forceinline__ void async_cp32(const f16* gptr, f16* lptr) {
  unsigned l = (unsigned)(size_t)lptr;  // low 32 bits of generic ptr == LDS byte addr
  asm volatile("global_load_async_to_lds_b128 %0, %1, off"
               :: "v"(l), "v"(gptr) : "memory");
  asm volatile("global_load_async_to_lds_b128 %0, %1, off offset:16"
               :: "v"(l), "v"(gptr) : "memory");
}
static __device__ __forceinline__ void wait_async0() {
  asm volatile("s_wait_asynccnt 0x0" ::: "memory");
}

// ---------------------------------------------------------------- convert
__global__ void cvt_kernel(const float* __restrict__ src, f16* __restrict__ dst, int n) {
  int i = blockIdx.x * blockDim.x + threadIdx.x;
  int stride = gridDim.x * blockDim.x;
  for (; i < n; i += stride) dst[i] = (f16)src[i];
}

// ---------------------------------------------------------------- GEMM: C = A * W^T
// A: (M x K) f16 row-major, W: (N x K) f16 row-major.
// mode 0: dstH f16 head-major [b][h][s][d]; mode 1: dstF f32 row-major (M x N)
__global__ __launch_bounds__(256)
void gemm_kernel(const f16* __restrict__ A, const f16* __restrict__ W,
                 int M, int N, int K, int mode, int Srows,
                 f16* __restrict__ dstH, float* __restrict__ dstF) {
  __shared__ __attribute__((aligned(32))) f16 As[2][128 * 32];
  __shared__ __attribute__((aligned(32))) f16 Bs[2][128 * 32];

  const int tid  = threadIdx.x;
  const int lane = tid & 31;
  const int wid  = tid >> 5;
  const int wm   = wid & 3;   // 4 wave rows  -> 32 M each
  const int wn   = wid >> 2;  // 2 wave cols  -> 64 N each
  const int m0   = blockIdx.y * 128;
  const int n0   = blockIdx.x * 128;

  const int ar = tid >> 1;        // 0..127 (tile row this thread stages)
  const int ac = (tid & 1) * 16;  // 0 or 16 (k sub-chunk)
  const int growA = m0 + ar;
  const int gnB   = n0 + ar;
  const f16* aRow = A + (size_t)growA * K + ac;
  const f16* bRow = W + (size_t)gnB * K + ac;

  auto stage = [&](int k0, int buf) {
    if (growA < M) {
      async_cp32(aRow + k0, &As[buf][ar * 32 + ac]);
    } else {
      v8h z = {};
      *(v8h*)&As[buf][ar * 32 + ac]     = z;
      *(v8h*)&As[buf][ar * 32 + ac + 8] = z;
    }
    if (gnB < N) {
      async_cp32(bRow + k0, &Bs[buf][ar * 32 + ac]);
    } else {
      v8h z = {};
      *(v8h*)&Bs[buf][ar * 32 + ac]     = z;
      *(v8h*)&Bs[buf][ar * 32 + ac + 8] = z;
    }
  };

  v8f acc[2][4] = {};

  stage(0, 0);
  wait_async0();
  __syncthreads();

  const int nk = K >> 5;
  for (int t = 0; t < nk; ++t) {
    const int buf = t & 1;
    if (t + 1 < nk) stage((t + 1) * 32, buf ^ 1);  // async prefetch next tile

    v16h af[2], bf[4];
#pragma unroll
    for (int mt = 0; mt < 2; ++mt) {
      const int r  = wm * 32 + mt * 16 + (lane & 15);
      const int ko = (lane < 16) ? 0 : 16;
      v8h lo = *(const v8h*)&As[buf][r * 32 + ko];
      v8h hi = *(const v8h*)&As[buf][r * 32 + ko + 8];
#pragma unroll
      for (int i = 0; i < 8; ++i) { af[mt][i] = lo[i]; af[mt][i + 8] = hi[i]; }
    }
#pragma unroll
    for (int nt = 0; nt < 4; ++nt) {
      const int c  = wn * 64 + nt * 16 + (lane & 15);
      const int ko = (lane < 16) ? 0 : 16;
      bf[nt] = *(const v16h*)&Bs[buf][c * 32 + ko];
    }
#pragma unroll
    for (int mt = 0; mt < 2; ++mt)
#pragma unroll
      for (int nt = 0; nt < 4; ++nt)
        acc[mt][nt] = wmma_f16(af[mt], bf[nt], acc[mt][nt]);

    wait_async0();      // next tile resident (this wave)
    __syncthreads();    // all waves done reading old buffer & staged new tile
  }

#pragma unroll
  for (int mt = 0; mt < 2; ++mt) {
#pragma unroll
    for (int nt = 0; nt < 4; ++nt) {
#pragma unroll
      for (int i = 0; i < 8; ++i) {
        const int cm = m0 + wm * 32 + mt * 16 + i + ((lane >= 16) ? 8 : 0);
        const int cn = n0 + wn * 64 + nt * 16 + (lane & 15);
        if (cm >= M) continue;
        const float val = acc[mt][nt][i];
        if (mode == 0) {
          const int b = cm / Srows, s = cm % Srows;
          const int h = cn >> 7,   d = cn & 127;
          dstH[(((size_t)(b * H_ + h)) * Srows + s) * HD_ + d] = (f16)val;
        } else {
          dstF[(size_t)cm * N + cn] = val;
        }
      }
    }
  }
}

// ---------------------------------------------------------------- RoPE (in-place on Q,K)
__global__ void rope_kernel(f16* __restrict__ q, f16* __restrict__ k,
                            const float* __restrict__ cosb, const float* __restrict__ sinb) {
  const int total = B_ * H_ * S_ * (HD_ / 2);
  int idx = blockIdx.x * blockDim.x + threadIdx.x;
  if (idx >= total) return;
  const int j  = idx & 63;
  const int s  = (idx >> 6) & (S_ - 1);
  const int hb = idx >> 17;  // b*H + h
  const float c  = cosb[s * 64 + j];
  const float sn = sinb[s * 64 + j];
  const size_t base = (((size_t)hb) * S_ + s) * HD_ + 2 * j;
  float q0 = (float)q[base], q1 = (float)q[base + 1];
  q[base]     = (f16)(q0 * c - q1 * sn);
  q[base + 1] = (f16)(q0 * sn + q1 * c);
  float k0 = (float)k[base], k1 = (float)k[base + 1];
  k[base]     = (f16)(k0 * c - k1 * sn);
  k[base + 1] = (f16)(k0 * sn + k1 * c);
}

// ---------------------------------------------------------------- flash attention + adapter
__global__ __launch_bounds__(256)
void attn_kernel(const f16* __restrict__ q, const f16* __restrict__ k,
                 const f16* __restrict__ v, const f16* __restrict__ ak,
                 const f16* __restrict__ av, const float* __restrict__ gate,
                 f16* __restrict__ out) {
  __shared__ __attribute__((aligned(32))) f16 Ks[32 * 128];   // [key][d]
  __shared__ __attribute__((aligned(32))) f16 Vt[128 * 32];   // [d][key]
  __shared__ __attribute__((aligned(32))) f16 AKs[16 * 128];  // adapter keys, padded
  __shared__ __attribute__((aligned(32))) f16 AVt[128 * 32];  // adapter V^T, padded
  __shared__ __attribute__((aligned(32))) f16 Ps[8][16 * 32]; // per-wave P scratch

  const int tid   = threadIdx.x;
  const int lane  = tid & 31;
  const int w     = tid >> 5;
  const int bh    = blockIdx.y;        // b*H + h
  const int h     = bh & (H_ - 1);
  const int chunk = blockIdx.x;        // 128 q rows per block
  const int q0    = chunk * 128 + w * 16;
  const float scale = 0.08838834764831845f;  // 1/sqrt(128)
  const float gt    = tanhf(gate[h]);
  const size_t bhS  = (size_t)bh * S_;

  // stage adapter K (async, zero-pad to 16 keys) and adapter V^T (padded to 32)
  if (tid < 128) {
    const int key = tid >> 3, dc = (tid & 7) * 16;
    if (key < AL_) {
      async_cp32(ak + (((size_t)bh) * AL_ + key) * HD_ + dc, &AKs[key * 128 + dc]);
    } else {
      v8h z = {};
      *(v8h*)&AKs[key * 128 + dc]     = z;
      *(v8h*)&AKs[key * 128 + dc + 8] = z;
    }
  }
  for (int i = tid; i < 128 * 32; i += 256) {
    const int d = i >> 5, key = i & 31;
    AVt[i] = (key < AL_) ? av[(((size_t)bh) * AL_ + key) * HD_ + d] : (f16)0.f;
  }

  // Q fragments (16 rows x 128 d -> 4 A-fragments), straight from global
  v16h qa[4];
  {
    const f16* qrow = q + (bhS + q0 + (lane & 15)) * HD_;
    const int ko = (lane < 16) ? 0 : 16;
#pragma unroll
    for (int kk = 0; kk < 4; ++kk) {
      v8h lo = *(const v8h*)(qrow + kk * 32 + ko);
      v8h hi = *(const v8h*)(qrow + kk * 32 + ko + 8);
#pragma unroll
      for (int i = 0; i < 8; ++i) { qa[kk][i] = lo[i]; qa[kk][i + 8] = hi[i]; }
    }
  }

  v8f o[8] = {};
  float mi[8], li[8];
#pragma unroll
  for (int i = 0; i < 8; ++i) { mi[i] = -1e30f; li[i] = 0.f; }

  const int nkb = (chunk + 1) * 4;  // causal: keys <= chunk*128+127
  for (int kb = 0; kb < nkb; ++kb) {
    const int kstart = kb * 32;
    {  // stage K (async 32x128) + transposed V (128x32 via ds stores)
      const int key = tid >> 3, dc = (tid & 7) * 16;
      async_cp32(k + (bhS + kstart + key) * HD_ + dc, &Ks[key * 128 + dc]);
      const f16* vp = v + (bhS + kstart + key) * HD_ + dc;
      v8h v0 = *(const v8h*)vp;
      v8h v1 = *(const v8h*)(vp + 8);
#pragma unroll
      for (int j = 0; j < 8; ++j) {
        Vt[(dc + j) * 32 + key]     = v0[j];
        Vt[(dc + 8 + j) * 32 + key] = v1[j];
      }
    }
    wait_async0();
    __syncthreads();

    if (q0 + 15 >= kstart) {
      const int ko = (lane < 16) ? 0 : 16;
      v8f s0 = {}, s1 = {};
#pragma unroll
      for (int kk = 0; kk < 4; ++kk) {
        v16h bf0 = *(const v16h*)&Ks[(0  + (lane & 15)) * 128 + kk * 32 + ko];
        v16h bf1 = *(const v16h*)&Ks[(16 + (lane & 15)) * 128 + kk * 32 + ko];
        s0 = wmma_f16(qa[kk], bf0, s0);
        s1 = wmma_f16(qa[kk], bf1, s1);
      }
      const int key0 = kstart + (lane & 15);
      const int key1 = key0 + 16;
#pragma unroll
      for (int i = 0; i < 8; ++i) {
        const int qr = q0 + i + ((lane >= 16) ? 8 : 0);
        float v0 = s0[i] * scale; if (key0 > qr) v0 = -1e30f;
        float v1 = s1[i] * scale; if (key1 > qr) v1 = -1e30f;
        float mx = fmaxf(v0, v1);
#pragma unroll
        for (int off = 8; off >= 1; off >>= 1) mx = fmaxf(mx, __shfl_xor(mx, off));
        const float mnew = fmaxf(mi[i], mx);
        const float p0 = __expf(v0 - mnew);
        const float p1 = __expf(v1 - mnew);
        float rs = p0 + p1;
#pragma unroll
        for (int off = 8; off >= 1; off >>= 1) rs += __shfl_xor(rs, off);
        const float corr = __expf(mi[i] - mnew);
        li[i] = li[i] * corr + rs;
        mi[i] = mnew;
#pragma unroll
        for (int nt = 0; nt < 8; ++nt) o[nt][i] *= corr;
        const int r = i + ((lane >= 16) ? 8 : 0);
        Ps[w][r * 32 + (lane & 15)]      = (f16)p0;
        Ps[w][r * 32 + 16 + (lane & 15)] = (f16)p1;
      }
      // P(16x32) @ V(32x128): C-layout -> A-layout via per-wave LDS scratch
      v16h pa;
      {
        const int r = lane & 15;
        const int k2 = (lane < 16) ? 0 : 16;
        v8h lo = *(const v8h*)&Ps[w][r * 32 + k2];
        v8h hi = *(const v8h*)&Ps[w][r * 32 + k2 + 8];
#pragma unroll
        for (int i = 0; i < 8; ++i) { pa[i] = lo[i]; pa[i + 8] = hi[i]; }
      }
#pragma unroll
      for (int nt = 0; nt < 8; ++nt) {
        v16h bf = *(const v16h*)&Vt[(nt * 16 + (lane & 15)) * 32 + ko];
        o[nt] = wmma_f16(pa, bf, o[nt]);
      }
    }
    __syncthreads();
  }

  // normalize main attention
#pragma unroll
  for (int nt = 0; nt < 8; ++nt)
#pragma unroll
    for (int i = 0; i < 8; ++i) o[nt][i] /= li[i];

  // adapter attention: separate softmax over 10 keys, scaled by tanh(gate)
  {
    const int ko = (lane < 16) ? 0 : 16;
    v8f sa = {};
#pragma unroll
    for (int kk = 0; kk < 4; ++kk) {
      v16h bf = *(const v16h*)&AKs[(lane & 15) * 128 + kk * 32 + ko];
      sa = wmma_f16(qa[kk], bf, sa);
    }
#pragma unroll
    for (int i = 0; i < 8; ++i) {
      float vv = sa[i] * scale;
      if ((lane & 15) >= AL_) vv = -1e30f;
      float mx = vv;
#pragma unroll
      for (int off = 8; off >= 1; off >>= 1) mx = fmaxf(mx, __shfl_xor(mx, off));
      const float p = __expf(vv - mx);
      float rs = p;
#pragma unroll
      for (int off = 8; off >= 1; off >>= 1) rs += __shfl_xor(rs, off);
      const int r = i + ((lane >= 16) ? 8 : 0);
      Ps[w][r * 32 + (lane & 15)]      = (f16)(p / rs * gt);
      Ps[w][r * 32 + 16 + (lane & 15)] = (f16)0.f;
    }
    v16h pa;
    {
      const int r = lane & 15;
      const int k2 = (lane < 16) ? 0 : 16;
      v8h lo = *(const v8h*)&Ps[w][r * 32 + k2];
      v8h hi = *(const v8h*)&Ps[w][r * 32 + k2 + 8];
#pragma unroll
      for (int i = 0; i < 8; ++i) { pa[i] = lo[i]; pa[i + 8] = hi[i]; }
    }
#pragma unroll
    for (int nt = 0; nt < 8; ++nt) {
      v16h bf = *(const v16h*)&AVt[(nt * 16 + (lane & 15)) * 32 + ko];
      o[nt] = wmma_f16(pa, bf, o[nt]);
    }
  }

  // store f16 row-major [b*S + s][h*HD + d] for the output projection
  {
    const int bb = bh >> 5;
#pragma unroll
    for (int nt = 0; nt < 8; ++nt) {
#pragma unroll
      for (int i = 0; i < 8; ++i) {
        const int r = i + ((lane >= 16) ? 8 : 0);
        const size_t row = (size_t)bb * S_ + q0 + r;
        const int col = h * HD_ + nt * 16 + (lane & 15);
        out[row * D_ + col] = (f16)o[nt][i];
      }
    }
  }
}

// ---------------------------------------------------------------- launch
extern "C" void kernel_launch(void* const* d_in, const int* in_sizes, int n_in,
                              void* d_out, int out_size, void* d_ws, size_t ws_size,
                              hipStream_t stream) {
  const float* x    = (const float*)d_in[0];
  const float* cosb = (const float*)d_in[2];
  const float* sinb = (const float*)d_in[3];
  const float* wq   = (const float*)d_in[4];
  const float* wk   = (const float*)d_in[5];
  const float* wv   = (const float*)d_in[6];
  const float* wo   = (const float*)d_in[7];
  const float* gate = (const float*)d_in[8];
  const float* adap = (const float*)d_in[9];
  float* out = (float*)d_out;
  (void)in_sizes; (void)n_in; (void)out_size; (void)ws_size;

  const size_t M = (size_t)B_ * S_;  // 4096
  char* ws = (char*)d_ws;
  size_t off = 0;
  auto alloc = [&](size_t bytes) {
    char* p = ws + off;
    off += (bytes + 255) & ~(size_t)255;
    return p;
  };
  f16* xh  = (f16*)alloc(M * D_ * 2);
  f16* wqh = (f16*)alloc((size_t)D_ * D_ * 2);
  f16* wkh = (f16*)alloc((size_t)D_ * D_ * 2);
  f16* wvh = (f16*)alloc((size_t)D_ * D_ * 2);
  f16* woh = (f16*)alloc((size_t)D_ * D_ * 2);
  f16* adh = (f16*)alloc((size_t)B_ * AL_ * D_ * 2);
  f16* qh  = (f16*)alloc(M * D_ * 2);
  f16* kh  = (f16*)alloc(M * D_ * 2);
  f16* vh  = (f16*)alloc(M * D_ * 2);
  f16* akh = (f16*)alloc((size_t)B_ * H_ * AL_ * HD_ * 2);
  f16* avh = (f16*)alloc((size_t)B_ * H_ * AL_ * HD_ * 2);
  f16* aoh = (f16*)alloc(M * D_ * 2);

  const int TB = 256;
  auto cvt = [&](const float* s, f16* d, size_t n) {
    int blocks = (int)((n + TB - 1) / TB);
    if (blocks > 65535) blocks = 65535;  // grid-stride
    cvt_kernel<<<blocks, TB, 0, stream>>>(s, d, (int)n);
  };
  cvt(x,    xh,  M * D_);
  cvt(wq,   wqh, (size_t)D_ * D_);
  cvt(wk,   wkh, (size_t)D_ * D_);
  cvt(wv,   wvh, (size_t)D_ * D_);
  cvt(wo,   woh, (size_t)D_ * D_);
  cvt(adap, adh, (size_t)B_ * AL_ * D_);

  const dim3 g(D_ / 128, (unsigned)((M + 127) / 128));
  gemm_kernel<<<g, 256, 0, stream>>>(xh, wqh, (int)M, D_, D_, 0, S_, qh, nullptr);
  gemm_kernel<<<g, 256, 0, stream>>>(xh, wkh, (int)M, D_, D_, 0, S_, kh, nullptr);
  gemm_kernel<<<g, 256, 0, stream>>>(xh, wvh, (int)M, D_, D_, 0, S_, vh, nullptr);

  const dim3 ga(D_ / 128, 1);
  gemm_kernel<<<ga, 256, 0, stream>>>(adh, wkh, B_ * AL_, D_, D_, 0, AL_, akh, nullptr);
  gemm_kernel<<<ga, 256, 0, stream>>>(adh, wvh, B_ * AL_, D_, D_, 0, AL_, avh, nullptr);

  rope_kernel<<<(B_ * H_ * S_ * (HD_ / 2)) / 256, 256, 0, stream>>>(qh, kh, cosb, sinb);

  attn_kernel<<<dim3(S_ / 128, B_ * H_), 256, 0, stream>>>(qh, kh, vh, akh, avh, gate, aoh);

  gemm_kernel<<<g, 256, 0, stream>>>(aoh, woh, (int)M, D_, D_, 1, S_, nullptr, out);
}
